// CrossModalAttention_23742579213046
// MI455X (gfx1250) — matrix operand order repaired and checked by reading, not compiled, not run on hardware
//
#include <hip/hip_runtime.h>
#include <hip/hip_bf16.h>
#include <cstdint>
#include <cstddef>

// ---------------------------------------------------------------------------
// CDNA5 (gfx1250) wave32 WMMA types
// ---------------------------------------------------------------------------
typedef __attribute__((ext_vector_type(16))) __bf16 v16bf;
typedef __attribute__((ext_vector_type(8)))  float  v8f;
typedef unsigned int u32x4 __attribute__((ext_vector_type(4)));
typedef int          i32x4 __attribute__((ext_vector_type(4)));
typedef int          i32x8 __attribute__((ext_vector_type(8)));

union FragU { uint32_t u[8]; uint4 q[2]; v16bf v; };
union AccU  { v8f v; float f[8]; };

#if defined(__AMDGCN__) && __has_builtin(__builtin_amdgcn_tensor_load_to_lds) && \
    __has_builtin(__builtin_amdgcn_s_wait_tensorcnt)
#define USE_TDM 1
#else
#define USE_TDM 0
#endif

__device__ __forceinline__ uint32_t f32_to_bf16_raw(float x) {
  union { float f; uint32_t u; } c; c.f = x;
  uint32_t u = c.u;
  return (u + 0x7FFFu + ((u >> 16) & 1u)) >> 16;   // round-to-nearest-even
}
__device__ __forceinline__ uint32_t pack_bf16(float lo, float hi) {
  return f32_to_bf16_raw(lo) | (f32_to_bf16_raw(hi) << 16);
}
__device__ __forceinline__ float bf16_lo(uint32_t w) {
  union { uint32_t u; float f; } c; c.u = w << 16; return c.f;
}
__device__ __forceinline__ float bf16_hi(uint32_t w) {
  union { uint32_t u; float f; } c; c.u = w & 0xFFFF0000u; return c.f;
}
__device__ __forceinline__ float bf16_val(uint16_t h) {
  union { uint32_t u; float f; } c; c.u = ((uint32_t)h) << 16; return c.f;
}

__device__ __forceinline__ float fast_tanh(float v) {
#if __has_builtin(__builtin_amdgcn_tanhf)
  return __builtin_amdgcn_tanhf(v);        // gfx1250 v_tanh_f32
#elif __has_builtin(__builtin_amdgcn_tanh_f32)
  return __builtin_amdgcn_tanh_f32(v);
#else
  return tanhf(v);
#endif
}

// Problem constants: D=256, H=4, HD=64, M=4, SCORE_H=64
#define QS 264   // bf16 row stride (mult of 8 -> 16B-aligned b128; 4-bank row stagger)
#define WS 264   // staged-weight row stride in LDS (same stagger; 528B = 512B + 16B pad)
#define HS 68    // f32 score-hidden row stride

// ---------------------------------------------------------------------------
// A-fragment (16x32 bf16 MxK), documented CDNA5 layout:
//  lanes 0-15: row M=lane, K={0..7,16..23}; lanes 16-31: K={8..15,24..31}
// ---------------------------------------------------------------------------
__device__ __forceinline__ v16bf load_afrag(const uint16_t* rowbase, int kt, int hi) {
  FragU a;
  a.q[0] = *(const uint4*)(rowbase + kt * 32 + hi * 8);
  a.q[1] = *(const uint4*)(rowbase + kt * 32 + hi * 8 + 16);
  return a.v;
}
// B-fragment (32x16 bf16 KxN) from LDS stage (32 E-rows, stride WS):
//  lanes 0-15: K={0..7,16..23}; lanes 16-31: K={8..15,24..31}.
__device__ __forceinline__ v16bf load_bfrag(const uint16_t* Wb, int n, int kt, int hi) {
  FragU b;
  const uint16_t* p = Wb + n * WS + kt * 32 + hi * 8;
  b.q[0] = *(const uint4*)(p);
  b.q[1] = *(const uint4*)(p + 16);
  return b.v;
}

// Fallback stage: 32 E-rows (32x256 bf16 = 16KB) -> LDS with padded stride WS.
__device__ __forceinline__ void stage32_copy(uint16_t* ldsW, const uint16_t* src, int tid) {
  const uint4* s = (const uint4*)src;
  uint4* d = (uint4*)ldsW;
  for (int i = tid; i < 1024; i += 128) {       // 16 uint4 per row of 256 u16
    int row = i >> 4, c = i & 15;
    d[row * 33 + c] = s[i];                     // 33 uint4 = 264 u16 per row
  }
}

// Stage one 32x256 bf16 weight tile into LDS.
// TDM path: single tensor_load_to_lds DMA; pad_enable reproduces the
// 16B-per-512B-row padding (row stride 528B = WS u16). Only wave 0 issues;
// TENSORcnt wait + the caller's __syncthreads publishes the tile.
__device__ __forceinline__ void stage_tile(uint16_t* ldsW, const uint16_t* src,
                                           int tid, int w) {
#if USE_TDM
  if (w == 0) {
    uint64_t ga  = (uint64_t)(uintptr_t)src;
    uint32_t lds = (uint32_t)(uintptr_t)ldsW;   // low 32 bits = LDS byte addr
    u32x4 g0;
    g0[0] = 1u;                                  // count=1, user mode, no gather
    g0[1] = lds;                                 // lds_addr [63:32]
    g0[2] = (uint32_t)ga;                        // global_addr [95:64]
    g0[3] = (uint32_t)((ga >> 32) & 0x01FFFFFFu) | 0x80000000u;  // [120:96] | type=2
    i32x8 g1;
    // data_size=2B (1<<16) | pad_enable (1<<20) | pad_interval=6:128dw (6<<22)
    // | pad_amount=3:4dw (3<<25)
    g1[0] = (int)0x07910000u;
    g1[1] = (int)(256u << 16);                   // tensor_dim0 = 256 (low 16 -> [31:16])
    g1[2] = (int)(32u << 16);                    // tensor_dim0 hi=0, tensor_dim1 = 32
    g1[3] = (int)(256u << 16);                   // tensor_dim1 hi=0, tile_dim0 = 256
    g1[4] = 32;                                  // tile_dim1 = 32, tile_dim2 = 0
    g1[5] = 256;                                 // tensor_dim0_stride = 256
    g1[6] = 0;                                   // stride0 hi, tensor_dim1_stride lo
    g1[7] = 0;
    i32x4 gz4 = {0, 0, 0, 0};                    // 2D tensor: groups 2/3 unused
#if __clang_major__ >= 23
    i32x8 gz8 = {0, 0, 0, 0, 0, 0, 0, 0};
    __builtin_amdgcn_tensor_load_to_lds(g0, g1, gz4, gz4, gz8, 0);
#else
    __builtin_amdgcn_tensor_load_to_lds(g0, g1, gz4, gz4, 0);
#endif
    __builtin_amdgcn_s_wait_tensorcnt(0);
  }
#else
  stage32_copy(ldsW, src, tid);
#endif
}

// ---------------------------------------------------------------------------
// Pre-kernel: fp32 weights -> bf16 workspace (row-major, L2-resident).
// Layout: [0)Wq [65536)Wk [131072)Wv [196608)Wo [262144)W1  (278528 elems)
// ---------------------------------------------------------------------------
__global__ void cma_cvt_weights(const float* __restrict__ Wq, const float* __restrict__ Wk,
                                const float* __restrict__ Wv, const float* __restrict__ Wo,
                                const float* __restrict__ W1, uint16_t* __restrict__ ws) {
  int i = blockIdx.x * 256 + threadIdx.x;
  if (i >= 278528) return;
  float v;
  if (i < 65536)        v = Wq[i];
  else if (i < 131072)  v = Wk[i - 65536];
  else if (i < 196608)  v = Wv[i - 131072];
  else if (i < 262144)  v = Wo[i - 196608];
  else                  v = W1[i - 262144];
  ws[i] = (uint16_t)f32_to_bf16_raw(v);
}

// ---------------------------------------------------------------------------
// Main fused kernel: 128 threads = 4 waves; wave = 4 batches (16 rows).
// 152.6KB LDS -> 2 workgroups per WGP for staging/scalar-phase overlap.
// ---------------------------------------------------------------------------
__global__ __launch_bounds__(128)
void cma_fuse_kernel(const float* __restrict__ x, const int* __restrict__ mask,
                     const uint16_t* __restrict__ wsW,
                     const float* __restrict__ b1, const float* __restrict__ W2,
                     const float* __restrict__ b2,
                     const float* __restrict__ bq, const float* __restrict__ bk,
                     const float* __restrict__ bv, const float* __restrict__ bo,
                     const float* __restrict__ gvec, const float* __restrict__ beta,
                     float* __restrict__ out, int B) {
  extern __shared__ char smem[];
  const int tid  = threadIdx.x;
  const int lane = tid & 31;
  const int w    = tid >> 5;
  const int nl   = lane & 15;
  const int hi   = lane >> 4;

  // --- LDS carve-up (total 152,576 B) ---
  uint16_t* ldsW  = (uint16_t*)smem;                       // 32*WS u16 = 16,896 B
  uint16_t* xbAll = ldsW + 32 * WS;                        // 4*16*QS bf16 each:
  uint16_t* qsAll = xbAll + 4 * 16 * QS;                   //   33,792 B x4
  uint16_t* ksAll = qsAll + 4 * 16 * QS;
  uint16_t* vsAll = ksAll + 4 * 16 * QS;
  float*    scAll = (float*)(vsAll + 4 * 16 * QS);         // 64 f32
  float*    alAll = scAll + 64;                            // 64 f32

  uint16_t* xb = xbAll + w * 16 * QS;
  uint16_t* qs = qsAll + w * 16 * QS;   // later reused as attn
  uint16_t* ks = ksAll + w * 16 * QS;
  uint16_t* vs = vsAll + w * 16 * QS;
  float*    hb = (float*)qs;            // overlay: dead before Q-projection
  float*    sc = scAll + w * 16;
  float*    al = alAll + w * 16;

  const int gbase = blockIdx.x * 16 + w * 4;

  // -------- Phase 1: stage x rows (contiguous) as bf16 --------------------
  const float* xg = x + (size_t)gbase * 1024;
#pragma unroll
  for (int t = 0; t < 32; ++t) {
    int i = (t * 32 + lane) * 4;                  // 0..4095
    int r = i >> 8, c = i & 255;
    float4 v4 = *(const float4*)(xg + i);
    uint2 pk; pk.x = pack_bf16(v4.x, v4.y); pk.y = pack_bf16(v4.z, v4.w);
    *(uint2*)(xb + r * QS + c) = pk;
  }

  // -------- Phase 2: score net h = tanh(x @ W1^T + b1) --------------------
  {
    AccU acc[4];
#pragma unroll
    for (int nt = 0; nt < 4; ++nt)
#pragma unroll
      for (int e = 0; e < 8; ++e) acc[nt].f[e] = 0.0f;
#pragma unroll
    for (int stg = 0; stg < 2; ++stg) {
      __syncthreads();
      stage_tile(ldsW, wsW + 262144 + stg * 8192, tid, w);
      __syncthreads();
      const uint16_t* arow = xb + nl * QS;
      for (int kt = 0; kt < 8; ++kt) {
        v16bf a = load_afrag(arow, kt, hi);
#pragma unroll
        for (int nt2 = 0; nt2 < 2; ++nt2) {
          v16bf bb = load_bfrag(ldsW, nt2 * 16 + nl, kt, hi);
          acc[stg * 2 + nt2].v = __builtin_amdgcn_wmma_f32_16x16x32_bf16(
              false, a, false, bb, (short)0, acc[stg * 2 + nt2].v, false, false);
        }
      }
    }
#pragma unroll
    for (int nt = 0; nt < 4; ++nt) {
      float bias = b1[nt * 16 + nl];
#pragma unroll
      for (int vv = 0; vv < 8; ++vv)
        hb[(vv + hi * 8) * HS + nt * 16 + nl] = fast_tanh(acc[nt].f[vv] + bias);
    }
  }
  __syncthreads();

  // -------- scores + masked softmax over modalities -> alpha --------------
  if (lane < 16) {
    float s = b2[0];
    for (int e = 0; e < 64; ++e) s += hb[lane * HS + e] * W2[e];
    sc[lane] = s;
  }
  __syncthreads();
  if (lane < 4) {
    int gb = gbase + lane;
    float sv[4]; int pres[4];
    float mx = -3.0e38f;
    for (int m = 0; m < 4; ++m) {
      pres[m] = (mask[gb * 4 + m] == 0);          // True = absent
      sv[m] = sc[4 * lane + m];
      if (pres[m] && sv[m] > mx) mx = sv[m];
    }
    float sum = 0.f, ev[4];
    for (int m = 0; m < 4; ++m) { ev[m] = pres[m] ? __expf(sv[m] - mx) : 0.f; sum += ev[m]; }
    float inv = 1.0f / sum;
    for (int m = 0; m < 4; ++m) al[4 * lane + m] = ev[m] * inv;
  }
  __syncthreads();

  // -------- Phase 3: Q/K/V projections, TDM-staged 16KB weight tiles ------
#pragma unroll
  for (int p = 0; p < 3; ++p) {
    const int    wbase = p * 65536;
    const float* bp    = (p == 0) ? bq : (p == 1) ? bk : bv;
    uint16_t*    dst   = (p == 0) ? qs : (p == 1) ? ks : vs;
    AccU acc[16];
#pragma unroll
    for (int nt = 0; nt < 16; ++nt)
#pragma unroll
      for (int e = 0; e < 8; ++e) acc[nt].f[e] = 0.0f;
#pragma unroll
    for (int stg = 0; stg < 8; ++stg) {
      __syncthreads();
      stage_tile(ldsW, wsW + wbase + stg * 8192, tid, w);
      __syncthreads();
      const uint16_t* arow = xb + nl * QS;
      for (int kt = 0; kt < 8; ++kt) {
        v16bf a = load_afrag(arow, kt, hi);
#pragma unroll
        for (int nt2 = 0; nt2 < 2; ++nt2) {
          v16bf bb = load_bfrag(ldsW, nt2 * 16 + nl, kt, hi);
          acc[stg * 2 + nt2].v = __builtin_amdgcn_wmma_f32_16x16x32_bf16(
              false, a, false, bb, (short)0, acc[stg * 2 + nt2].v, false, false);
        }
      }
    }
#pragma unroll
    for (int nt = 0; nt < 16; ++nt) {
      float bias = bp[nt * 16 + nl];
#pragma unroll
      for (int vv = 0; vv < 8; ++vv)
        dst[(vv + hi * 8) * QS + nt * 16 + nl] =
            (uint16_t)f32_to_bf16_raw(acc[nt].f[vv] + bias);
    }
  }
  __syncthreads();

  // -------- Phase 4: per-head attention (4x4 per head), packed bf16 -------
  // attn overwrites this lane's own q block (reads precede writes in order).
  if (lane < 16) {
    int b = lane >> 2, h = lane & 3;
    int gb = gbase + b;
    int r0 = 4 * b, cb = h * 64;
    float p[4][4];
    for (int qm = 0; qm < 4; ++qm) {
      float lg[4];
      const uint16_t* qp = qs + (r0 + qm) * QS + cb;
      for (int km = 0; km < 4; ++km) {
        const uint16_t* kp = ks + (r0 + km) * QS + cb;
        float s = 0.f;
        for (int t = 0; t < 64; t += 2) {
          uint32_t qw = *(const uint32_t*)(qp + t);
          uint32_t kw = *(const uint32_t*)(kp + t);
          s += bf16_lo(qw) * bf16_lo(kw) + bf16_hi(qw) * bf16_hi(kw);
        }
        lg[km] = s * 0.125f;                      // 1/sqrt(64)
      }
      float mx = -3.0e38f;
      for (int km = 0; km < 4; ++km)
        if (mask[gb * 4 + km] == 0 && lg[km] > mx) mx = lg[km];
      float sum = 0.f;
      for (int km = 0; km < 4; ++km) {
        float e = (mask[gb * 4 + km] == 0) ? __expf(lg[km] - mx) : 0.f;
        p[qm][km] = e; sum += e;
      }
      float inv = 1.f / sum;
      for (int km = 0; km < 4; ++km) p[qm][km] *= inv;
    }
    for (int qm = 0; qm < 4; ++qm) {
      float p0 = p[qm][0], p1 = p[qm][1], p2 = p[qm][2], p3 = p[qm][3];
      for (int t = 0; t < 64; t += 2) {
        uint32_t v0 = *(const uint32_t*)(vs + (r0 + 0) * QS + cb + t);
        uint32_t v1 = *(const uint32_t*)(vs + (r0 + 1) * QS + cb + t);
        uint32_t v2 = *(const uint32_t*)(vs + (r0 + 2) * QS + cb + t);
        uint32_t v3 = *(const uint32_t*)(vs + (r0 + 3) * QS + cb + t);
        float s0 = p0 * bf16_lo(v0) + p1 * bf16_lo(v1) + p2 * bf16_lo(v2) + p3 * bf16_lo(v3);
        float s1 = p0 * bf16_hi(v0) + p1 * bf16_hi(v1) + p2 * bf16_hi(v2) + p3 * bf16_hi(v3);
        *(uint32_t*)(qs + (r0 + qm) * QS + cb + t) = pack_bf16(s0, s1);
      }
    }
  }

  // -------- Phase 5: out-proj + in-register LayerNorm + fusion ------------
  {
    AccU acc[16];
#pragma unroll
    for (int nt = 0; nt < 16; ++nt)
#pragma unroll
      for (int e = 0; e < 8; ++e) acc[nt].f[e] = 0.0f;
#pragma unroll
    for (int stg = 0; stg < 8; ++stg) {
      __syncthreads();
      stage_tile(ldsW, wsW + 196608 + stg * 8192, tid, w);
      __syncthreads();
      const uint16_t* arow = qs + nl * QS;        // attn (bf16)
      for (int kt = 0; kt < 8; ++kt) {
        v16bf a = load_afrag(arow, kt, hi);
#pragma unroll
        for (int nt2 = 0; nt2 < 2; ++nt2) {
          v16bf bb = load_bfrag(ldsW, nt2 * 16 + nl, kt, hi);
          acc[stg * 2 + nt2].v = __builtin_amdgcn_wmma_f32_16x16x32_bf16(
              false, a, false, bb, (short)0, acc[stg * 2 + nt2].v, false, false);
        }
      }
    }

    // alphas for this lane's 8 rows (row M = vv + hi*8; al[M] = alpha[b][m])
    float av[8];
#pragma unroll
    for (int vv = 0; vv < 8; ++vv) av[vv] = al[hi * 8 + vv];

    // y = x + attn_out + bo; also accumulate fused_scalar = sum alpha*x
    float fu0[16], fu1[16];
#pragma unroll
    for (int nt = 0; nt < 16; ++nt) {
      int N = nt * 16 + nl;
      float bias = bo[N];
      float s0 = 0.f, s1 = 0.f;
#pragma unroll
      for (int vv = 0; vv < 8; ++vv) {
        float xv = bf16_val(xb[(vv + hi * 8) * QS + N]);
        acc[nt].f[vv] += bias + xv;
        float t = av[vv] * xv;
        if (vv < 4) s0 += t; else s1 += t;
      }
      fu0[nt] = s0; fu1[nt] = s1;
    }

    // LayerNorm stats per row via 16-lane xor-shuffle reductions
    float mu[8], rs[8];
#pragma unroll
    for (int vv = 0; vv < 8; ++vv) {
      float s = 0.f;
#pragma unroll
      for (int nt = 0; nt < 16; ++nt) s += acc[nt].f[vv];
      s += __shfl_xor(s, 1, 32); s += __shfl_xor(s, 2, 32);
      s += __shfl_xor(s, 4, 32); s += __shfl_xor(s, 8, 32);
      mu[vv] = s * (1.0f / 256.0f);
      float v2 = 0.f;
#pragma unroll
      for (int nt = 0; nt < 16; ++nt) {
        float t = acc[nt].f[vv] - mu[vv]; v2 += t * t;
      }
      v2 += __shfl_xor(v2, 1, 32); v2 += __shfl_xor(v2, 2, 32);
      v2 += __shfl_xor(v2, 4, 32); v2 += __shfl_xor(v2, 8, 32);
      rs[vv] = rsqrtf(v2 * (1.0f / 256.0f) + 1e-5f);
    }

    // normed + alpha-weighted fusion; lane covers 2 batches of its hi half
#pragma unroll
    for (int nt = 0; nt < 16; ++nt) {
      int N = nt * 16 + nl;
      float g = gvec[N], be = beta[N];
      float f0 = fu0[nt], f1 = fu1[nt];
#pragma unroll
      for (int vv = 0; vv < 8; ++vv) {
        float nrm = (acc[nt].f[vv] - mu[vv]) * rs[vv] * g + be;
        float t = av[vv] * nrm;
        if (vv < 4) f0 += t; else f1 += t;
      }
      out[(size_t)(gbase + hi * 2 + 0) * 256 + N] = f0;
      out[(size_t)(gbase + hi * 2 + 1) * 256 + N] = f1;
    }
  }

  if (lane < 16) {
    int b = lane >> 2, m = lane & 3;
    out[(size_t)B * 256 + (size_t)(gbase + b) * 4 + m] = al[lane];
  }
}

// ---------------------------------------------------------------------------
extern "C" void kernel_launch(void* const* d_in, const int* in_sizes, int n_in,
                              void* d_out, int out_size, void* d_ws, size_t ws_size,
                              hipStream_t stream) {
  const float* x    = (const float*)d_in[0];
  const int*   mask = (const int*)d_in[1];
  const float* W1   = (const float*)d_in[2];
  const float* b1   = (const float*)d_in[3];
  const float* W2   = (const float*)d_in[4];
  const float* b2   = (const float*)d_in[5];
  const float* Wq   = (const float*)d_in[6];
  const float* bq   = (const float*)d_in[7];
  const float* Wk   = (const float*)d_in[8];
  const float* bk   = (const float*)d_in[9];
  const float* Wv   = (const float*)d_in[10];
  const float* bv   = (const float*)d_in[11];
  const float* Wo   = (const float*)d_in[12];
  const float* bo   = (const float*)d_in[13];
  const float* gvec = (const float*)d_in[14];
  const float* beta = (const float*)d_in[15];

  int B = in_sizes[0] / 1024;                     // B*M*D / (4*256)
  uint16_t* wsW = (uint16_t*)d_ws;

  cma_cvt_weights<<<(278528 + 255) / 256, 256, 0, stream>>>(Wq, Wk, Wv, Wo, W1, wsW);

  size_t shmem = (size_t)(32 * WS) * 2            // weight stage (16,896 B)
               + (size_t)4 * (4 * 16 * QS) * 2    // xb + qs + ks + vs
               + 2 * 64 * 4;                      // sc, al
  cma_fuse_kernel<<<B / 16, 128, shmem, stream>>>(
      x, mask, wsW, b1, W2, b2, bq, bk, bv, bo, gvec, beta, (float*)d_out, B);
}
